// ChatGLM2Attention_19086834663600
// MI455X (gfx1250) — compile-verified, matrix operand order checked by reading, not run on hardware
//
#include <hip/hip_runtime.h>
#include <hip/hip_bf16.h>

// ---------------------------------------------------------------------------
// CDNA5 (gfx1250) ChatGLM2 attention: bf16 WMMA GEMMs + flash attention.
// ---------------------------------------------------------------------------

typedef __attribute__((ext_vector_type(16))) __bf16 v16bf;
typedef __attribute__((ext_vector_type(8)))  float  v8f;

#define SEQ 2048
#define HID 4096
#define QKV_OUT 4608          // 4096 + 2*2*128
#define NH 32
#define KVH 2
#define HD 128

// ---------------------------------------------------------------------------
// fp32 -> bf16 elementwise convert
// ---------------------------------------------------------------------------
__global__ __launch_bounds__(256) void f32_to_bf16_kernel(const float* __restrict__ in,
                                                          __bf16* __restrict__ out,
                                                          long long n) {
  long long i = (long long)blockIdx.x * 256 + threadIdx.x;
  if (i < n) out[i] = (__bf16)in[i];
}

// ---------------------------------------------------------------------------
// C[M,N] = A[M,K] * W[N,K]^T + bias   (bf16 in, fp32 accumulate via WMMA)
// block 256 threads = 8 waves. Block tile 128x128, wave tile 32x64 (2x4 WMMA).
// ---------------------------------------------------------------------------
#define TM 128
#define TN 128
#define TK 32
#define LPAD 16   // bf16 elems of row padding (row stride 96B, 32B aligned)

__global__ __launch_bounds__(256)
void gemm_bf16_wmma(const __bf16* __restrict__ A, const __bf16* __restrict__ W,
                    const float* __restrict__ bias, void* __restrict__ Cout,
                    int M, int N, int K, int out_f32) {
  __shared__ __attribute__((aligned(32))) __bf16 lA[TM][TK + LPAD];
  __shared__ __attribute__((aligned(32))) __bf16 lB[TN][TK + LPAD];

  const int tid  = threadIdx.x;
  const int wave = tid >> 5;
  const int lane = tid & 31;
  const int wm   = wave >> 1;          // 0..3
  const int wn   = wave & 1;           // 0..1
  const int half = lane >> 4;          // 0/1 -> K 0..15 / 16..31
  const int lr   = lane & 15;

  const int m0 = blockIdx.y * TM;
  const int n0 = blockIdx.x * TN;

  // staging: each thread copies 32B of the A tile and 32B of the B tile
  const int srow = tid >> 1;           // 0..127
  const int sseg = (tid & 1) * 16;     // bf16 elems

  v8f acc[2][4] = {};

  for (int k0 = 0; k0 < K; k0 += TK) {
    const __bf16* gA = A + (size_t)(m0 + srow) * K + k0 + sseg;
    const __bf16* gW = W + (size_t)(n0 + srow) * K + k0 + sseg;
    uint4 a0 = ((const uint4*)gA)[0];
    uint4 a1 = ((const uint4*)gA)[1];
    uint4 b0 = ((const uint4*)gW)[0];
    uint4 b1 = ((const uint4*)gW)[1];
    if (k0 + TK < K) {                       // prefetch next K tile into caches
      __builtin_prefetch(gA + TK, 0, 1);
      __builtin_prefetch(gW + TK, 0, 1);
    }
    *(uint4*)&lA[srow][sseg]     = a0;
    *(uint4*)&lA[srow][sseg + 8] = a1;
    *(uint4*)&lB[srow][sseg]     = b0;
    *(uint4*)&lB[srow][sseg + 8] = b1;
    __syncthreads();

    v16bf afr[2], bfr[4];
#pragma unroll
    for (int tm = 0; tm < 2; tm++)
      afr[tm] = *(const v16bf*)&lA[wm * 32 + tm * 16 + lr][half * 16];
#pragma unroll
    for (int tn = 0; tn < 4; tn++)
      bfr[tn] = *(const v16bf*)&lB[wn * 64 + tn * 16 + lr][half * 16];

#pragma unroll
    for (int tm = 0; tm < 2; tm++)
#pragma unroll
      for (int tn = 0; tn < 4; tn++)
        acc[tm][tn] = __builtin_amdgcn_wmma_f32_16x16x32_bf16(
            false, afr[tm], false, bfr[tn], (short)0, acc[tm][tn], false, false);
    __syncthreads();
  }

  // epilogue: C-layout -> memory. lane<16: M=v, lane>=16: M=8+v; N = lane&15.
#pragma unroll
  for (int tm = 0; tm < 2; tm++)
#pragma unroll
    for (int tn = 0; tn < 4; tn++) {
      const int n  = n0 + wn * 64 + tn * 16 + lr;
      const float bv = bias ? bias[n] : 0.0f;
#pragma unroll
      for (int v = 0; v < 8; v++) {
        const int m = m0 + wm * 32 + tm * 16 + half * 8 + v;
        const float val = acc[tm][tn][v] + bv;
        if (out_f32) ((float*)Cout)[(size_t)m * N + n] = val;
        else         ((__bf16*)Cout)[(size_t)m * N + n] = (__bf16)val;
      }
    }
}

// ---------------------------------------------------------------------------
// RoPE on q/k (interleaved-pair variant), head-major bf16 outputs.
// q additionally pre-scaled by 1/sqrt(HD).  grid (B*S, NH+KVH), block 128.
// ---------------------------------------------------------------------------
__global__ __launch_bounds__(128)
void rope_kernel(const __bf16* __restrict__ qkv, __bf16* __restrict__ qout,
                 __bf16* __restrict__ kout) {
  const int bs = blockIdx.x;
  const int b  = bs / SEQ;
  const int s  = bs % SEQ;
  const int h  = blockIdx.y;             // 0..31 q heads, 32..33 k heads
  const int d  = threadIdx.x;            // 0..127

  const int col = (h < NH) ? h * HD + d : HID + (h - NH) * HD + d;
  float x = (float)qkv[(size_t)bs * QKV_OUT + col];
  float o;
  if (d < 64) {
    const int p = d >> 1;
    // inv_freq = 10000^(-p/32) = exp(-p * ln(10000)/32)
    const float inv = __expf(-(float)p * 0.28782313663f);
    float c, sn;
    __sincosf((float)s * inv, &c, &sn);
    const float xo = (float)qkv[(size_t)bs * QKV_OUT + (col ^ 1)];
    o = (d & 1) ? (x * c + xo * sn) : (x * c - xo * sn);
  } else {
    o = x;
  }
  if (h < NH) {
    o *= 0.08838834764831845f;  // 1/sqrt(128) folded into Q
    qout[(((size_t)b * NH + h) * SEQ + s) * HD + d] = (__bf16)o;
  } else {
    kout[(((size_t)b * KVH + (h - NH)) * SEQ + s) * HD + d] = (__bf16)o;
  }
}

// ---------------------------------------------------------------------------
// V transpose: vt[b][kv][d][s] = qkv[b][s][4352 + kv*128 + d]  (bf16)
// grid (SEQ/256, HD, B*KVH), block 256
// ---------------------------------------------------------------------------
__global__ __launch_bounds__(256)
void transpose_v_kernel(const __bf16* __restrict__ qkv, __bf16* __restrict__ vt) {
  const int s  = blockIdx.x * 256 + threadIdx.x;
  const int d  = blockIdx.y;
  const int bk = blockIdx.z;
  const __bf16 v = qkv[((size_t)(bk >> 1) * SEQ + s) * QKV_OUT +
                       (HID + KVH * HD) + (bk & 1) * HD + d];
  vt[((size_t)bk * HD + d) * SEQ + s] = v;
}

// ---------------------------------------------------------------------------
// Flash attention (causal, GQA rep=16). One wave per 16-row query block.
// grid (SEQ/64, NH, B), block 128 (4 waves). No block-wide barriers (waves
// run divergent key-loop lengths); P transpose goes through per-wave LDS with
// an explicit s_wait_dscnt for wave-local DS ordering.
// ---------------------------------------------------------------------------
__global__ __launch_bounds__(128)
void flash_attn_kernel(const __bf16* __restrict__ Q, const __bf16* __restrict__ Kb,
                       const __bf16* __restrict__ Vt, __bf16* __restrict__ ctx) {
  __shared__ __attribute__((aligned(32))) __bf16 pshare[4][16][32];

  const int tid  = threadIdx.x;
  const int wave = tid >> 5;
  const int lane = tid & 31;
  const int half = lane >> 4;
  const int lr   = lane & 15;
  const int h    = blockIdx.y;
  const int b    = blockIdx.z;
  const int kvh  = h >> 4;                       // NH/KVH = 16
  const int q0   = blockIdx.x * 64 + wave * 16;

  const __bf16* Qh = Q  + ((size_t)b * NH  + h)   * SEQ * HD;
  const __bf16* Kh = Kb + ((size_t)b * KVH + kvh) * SEQ * HD;
  const __bf16* Vh = Vt + ((size_t)b * KVH + kvh) * HD * SEQ;
  __bf16* pw = &pshare[wave][0][0];

  // Q fragments: 16 rows x 128 d = 4 A-operands (K=32 each), kept in regs
  v16bf qf[4];
#pragma unroll
  for (int c = 0; c < 4; c++)
    qf[c] = *(const v16bf*)(Qh + (size_t)(q0 + lr) * HD + c * 32 + half * 16);

  v8f o[8] = {};
  float mrow[8], lrow[8];
#pragma unroll
  for (int v = 0; v < 8; v++) { mrow[v] = -1.0e30f; lrow[v] = 0.0f; }

  const int nkb = (q0 + 16 + 31) >> 5;           // key blocks of 32
  for (int kb = 0; kb < nkb; kb++) {
    const int k0 = kb * 32;

    // ---- scores: S = Q * K^T  (two 16x16 n-tiles over 32 keys) ----
    v8f st[2] = {};
#pragma unroll
    for (int nt = 0; nt < 2; nt++)
#pragma unroll
      for (int c = 0; c < 4; c++) {
        v16bf kf = *(const v16bf*)(Kh + (size_t)(k0 + nt * 16 + lr) * HD +
                                   c * 32 + half * 16);
        st[nt] = __builtin_amdgcn_wmma_f32_16x16x32_bf16(
            false, qf[c], false, kf, (short)0, st[nt], false, false);
      }

    // ---- causal mask (diagonal blocks only) ----
    if (k0 + 31 > q0) {
#pragma unroll
      for (int nt = 0; nt < 2; nt++) {
        const int cgl = k0 + nt * 16 + lr;
#pragma unroll
        for (int v = 0; v < 8; v++)
          if (cgl > q0 + half * 8 + v) st[nt][v] = -3.0e38f;
      }
    }

    // ---- online softmax: row max / rescale / exp / row sum ----
    float alpha[8];
#pragma unroll
    for (int v = 0; v < 8; v++) {
      float mx = fmaxf(st[0][v], st[1][v]);
#pragma unroll
      for (int off = 8; off >= 1; off >>= 1)
        mx = fmaxf(mx, __shfl_xor(mx, off, 32));
      const float mn = fmaxf(mrow[v], mx);
      alpha[v] = __expf(mrow[v] - mn);
      mrow[v]  = mn;
    }
#pragma unroll
    for (int nt = 0; nt < 2; nt++)
#pragma unroll
      for (int v = 0; v < 8; v++)
        st[nt][v] = __expf(st[nt][v] - mrow[v]);
#pragma unroll
    for (int v = 0; v < 8; v++) {
      float rs = st[0][v] + st[1][v];
#pragma unroll
      for (int off = 8; off >= 1; off >>= 1)
        rs += __shfl_xor(rs, off, 32);
      lrow[v] = lrow[v] * alpha[v] + rs;
    }
#pragma unroll
    for (int dt = 0; dt < 8; dt++)
#pragma unroll
      for (int v = 0; v < 8; v++)
        o[dt][v] *= alpha[v];

    // ---- P: C-layout -> A-operand layout via per-wave LDS ----
#pragma unroll
    for (int nt = 0; nt < 2; nt++)
#pragma unroll
      for (int v = 0; v < 8; v++)
        pw[(half * 8 + v) * 32 + nt * 16 + lr] = (__bf16)st[nt][v];
    asm volatile("s_wait_dscnt 0" ::: "memory");   // wave-local DS RAW fence
    v16bf pf = *(const v16bf*)(pw + lr * 32 + half * 16);

    // ---- O += P * V  (V pre-transposed: B-operand contiguous along keys) ----
#pragma unroll
    for (int dt = 0; dt < 8; dt++) {
      v16bf vf = *(const v16bf*)(Vh + (size_t)(dt * 16 + lr) * SEQ +
                                 k0 + half * 16);
      o[dt] = __builtin_amdgcn_wmma_f32_16x16x32_bf16(
          false, pf, false, vf, (short)0, o[dt], false, false);
    }
  }

  // ---- normalize and store ctx[b][s][h*128+d] as bf16 ----
#pragma unroll
  for (int dt = 0; dt < 8; dt++)
#pragma unroll
    for (int v = 0; v < 8; v++) {
      const int srow = q0 + half * 8 + v;
      const float val = o[dt][v] / lrow[v];
      ctx[((size_t)b * SEQ + srow) * (NH * HD) + h * HD + dt * 16 + lr] =
          (__bf16)val;
    }
}

// ---------------------------------------------------------------------------
// launch
// ---------------------------------------------------------------------------
extern "C" void kernel_launch(void* const* d_in, const int* in_sizes, int n_in,
                              void* d_out, int out_size, void* d_ws, size_t ws_size,
                              hipStream_t stream) {
  const float* hidden  = (const float*)d_in[0];   // [2,2048,4096]
  const float* qkv_w   = (const float*)d_in[1];   // [4608,4096]
  const float* qkv_b   = (const float*)d_in[2];   // [4608]
  const float* dense_w = (const float*)d_in[3];   // [4096,4096]
  float* out = (float*)d_out;                     // [2,2048,4096]

  char* ws = (char*)d_ws;
  const size_t SZ_HIDA = (size_t)HID * (2 * SEQ) * 2;        // 32 MB bf16
  const size_t SZ_WQ   = (size_t)QKV_OUT * HID * 2;          // 36 MB
  const size_t SZ_WD   = (size_t)HID * HID * 2;              // 32 MB
  const size_t SZ_QKV  = (size_t)(2 * SEQ) * QKV_OUT * 2;    // 36 MB
  const size_t SZ_Q    = (size_t)2 * NH * SEQ * HD * 2;      // 32 MB
  const size_t SZ_K    = (size_t)2 * KVH * SEQ * HD * 2;     // 2 MB

  __bf16* hidA = (__bf16*)(ws);
  __bf16* wq   = (__bf16*)(ws + SZ_HIDA);
  __bf16* wd   = (__bf16*)(ws + SZ_HIDA + SZ_WQ);
  __bf16* qkvb = (__bf16*)(ws + SZ_HIDA + SZ_WQ + SZ_WD);
  __bf16* qbuf = (__bf16*)(ws + SZ_HIDA + SZ_WQ + SZ_WD + SZ_QKV);
  __bf16* kbuf = (__bf16*)(ws + SZ_HIDA + SZ_WQ + SZ_WD + SZ_QKV + SZ_Q);
  __bf16* vt   = (__bf16*)(ws + SZ_HIDA + SZ_WQ + SZ_WD + SZ_QKV + SZ_Q + SZ_K);
  __bf16* ctxb = hidA;   // hidden bf16 dead after gemm1; reuse for ctx

  const long long n_hid = (long long)(2 * SEQ) * HID;
  const long long n_wq  = (long long)QKV_OUT * HID;
  const long long n_wd  = (long long)HID * HID;

  f32_to_bf16_kernel<<<(unsigned)(n_hid / 256), 256, 0, stream>>>(hidden, hidA, n_hid);
  f32_to_bf16_kernel<<<(unsigned)(n_wq  / 256), 256, 0, stream>>>(qkv_w,  wq,  n_wq);
  f32_to_bf16_kernel<<<(unsigned)(n_wd  / 256), 256, 0, stream>>>(dense_w, wd, n_wd);

  // QKV projection: [4096,4608] bf16 out
  gemm_bf16_wmma<<<dim3(QKV_OUT / TN, (2 * SEQ) / TM), 256, 0, stream>>>(
      hidA, wq, qkv_b, qkvb, 2 * SEQ, QKV_OUT, HID, 0);

  // RoPE (q pre-scaled) + V transpose
  rope_kernel<<<dim3(2 * SEQ, NH + KVH), 128, 0, stream>>>(qkvb, qbuf, kbuf);
  transpose_v_kernel<<<dim3(SEQ / 256, HD, 2 * KVH), 256, 0, stream>>>(qkvb, vt);

  // causal flash attention -> ctx bf16 [4096, 4096]
  flash_attn_kernel<<<dim3(SEQ / 64, NH, 2), 128, 0, stream>>>(qbuf, kbuf, vt, ctxb);

  // output projection: fp32 out
  gemm_bf16_wmma<<<dim3(HID / TN, (2 * SEQ) / TM), 256, 0, stream>>>(
      ctxb, wd, nullptr, out, 2 * SEQ, HID, HID, 1);
}
